// GraphMultiHeadAttention_17377437679668
// MI455X (gfx1250) — matrix operand order repaired and checked by reading, not compiled, not run on hardware
//
#include <hip/hip_runtime.h>
#include <hip/hip_bf16.h>

// GraphMultiHeadAttention for MI455X (gfx1250, wave32, WMMA + TDM).
// B=2, N=1024, E=768, H=12, D=64.
//
// Roofline: key_edge_features = 512MB stream dominates (22us @ 23.3TB/s).
// Strategy: edge tile DMA'd to LDS via Tensor Data Mover (double-buffered,
// TENSORcnt-tracked) ONCE per (b, n-block), shared by all 12 heads; scores
// never hit HBM (online softmax); all GEMM-shaped math on bf16 WMMA.

#define BB 2
#define NN 1024
#define EE 768
#define HH 12
#define DD 64

typedef __attribute__((ext_vector_type(16))) __bf16 v16bf;
typedef __attribute__((ext_vector_type(8)))  float  v8f;
typedef unsigned int u32;
typedef unsigned long long u64;
typedef __attribute__((ext_vector_type(4))) u32 v4u;
typedef __attribute__((ext_vector_type(8))) int v8i;
typedef __attribute__((ext_vector_type(4))) int v4i;

// ---------------------------------------------------------------------------
// Generic  out = X @ W^T + bias  GEMM on bf16 WMMA (M=2048, N=768, K=768).
// mode != 3: write bf16 head-major [b][h][n][d] (outB), optionally fp32 too.
// mode == 3: write fp32 flat [r][c] (output projection -> d_out).
// ---------------------------------------------------------------------------
__global__ void __launch_bounds__(256)
wmma_proj_kernel(const float* __restrict__ X, const float* __restrict__ W,
                 const float* __restrict__ bias, int mode,
                 __bf16* __restrict__ outB, float* __restrict__ outF)
{
    __shared__ __bf16 As[64][40];   // pad 40 -> 80B rows: 16B-aligned, conflict-free frags
    __shared__ __bf16 Bs[64][40];

    const int tid  = threadIdx.x;
    const int lane = tid & 31;
    const int wave = tid >> 5;          // 8 waves: 4 (M) x 2 (N)
    const int wm   = wave & 3;
    const int wn   = wave >> 2;
    const int half = lane >> 4;
    const int lc   = lane & 15;

    const int rowBase = blockIdx.x * 64;
    const int colBase = blockIdx.y * 64;

    v8f acc0 = {};
    v8f acc1 = {};

    for (int kb = 0; kb < EE; kb += 32) {
        // ---- stage fp32 -> bf16 tiles into LDS (256 thr * 8 elem) ----
        {
            const int r  = tid >> 2;
            const int kc = (tid & 3) * 8;
            const float* xs = X + (size_t)(rowBase + r) * EE + kb + kc;
            const float* ws = W + (size_t)(colBase + r) * EE + kb + kc;
            float4 x0 = *(const float4*)(xs);
            float4 x1 = *(const float4*)(xs + 4);
            float4 w0 = *(const float4*)(ws);
            float4 w1 = *(const float4*)(ws + 4);
            As[r][kc + 0] = (__bf16)x0.x; As[r][kc + 1] = (__bf16)x0.y;
            As[r][kc + 2] = (__bf16)x0.z; As[r][kc + 3] = (__bf16)x0.w;
            As[r][kc + 4] = (__bf16)x1.x; As[r][kc + 5] = (__bf16)x1.y;
            As[r][kc + 6] = (__bf16)x1.z; As[r][kc + 7] = (__bf16)x1.w;
            Bs[r][kc + 0] = (__bf16)w0.x; Bs[r][kc + 1] = (__bf16)w0.y;
            Bs[r][kc + 2] = (__bf16)w0.z; Bs[r][kc + 3] = (__bf16)w0.w;
            Bs[r][kc + 4] = (__bf16)w1.x; Bs[r][kc + 5] = (__bf16)w1.y;
            Bs[r][kc + 6] = (__bf16)w1.z; Bs[r][kc + 7] = (__bf16)w1.w;
        }
        __syncthreads();

        // ---- build fragments per documented CDNA5 layouts ----
        v16bf a, b0, b1;
        const int arow = wm * 16 + lc;           // A: lane holds row M = lane%16
        #pragma unroll
        for (int j = 0; j < 8; ++j) {
            const int k0 = (j < 4) ? (2 * j + 8 * half) : (16 + 2 * (j - 4) + 8 * half);
            a[2 * j]     = As[arow][k0];
            a[2 * j + 1] = As[arow][k0 + 1];
        }
        const int c0 = wn * 32 + lc;             // B: lane holds col N = lane%16
        const int c1 = c0 + 16;
        #pragma unroll
        for (int j = 0; j < 8; ++j) {
            const int k = (j < 4) ? (2 * j + 8 * half) : (16 + 2 * (j - 4) + 8 * half);
            b0[2 * j]     = Bs[c0][k];
            b0[2 * j + 1] = Bs[c0][k + 1];
            b1[2 * j]     = Bs[c1][k];
            b1[2 * j + 1] = Bs[c1][k + 1];
        }

        acc0 = __builtin_amdgcn_wmma_f32_16x16x32_bf16(false, a, false, b0, (short)0, acc0, false, false);
        acc1 = __builtin_amdgcn_wmma_f32_16x16x32_bf16(false, a, false, b1, (short)0, acc1, false, false);
        __syncthreads();
    }

    // ---- epilogue: bias add + store (C layout: row = j + 8*half, col = lane%16) ----
    const int rowW = rowBase + wm * 16;
    #pragma unroll
    for (int t = 0; t < 2; ++t) {
        v8f& acc = t ? acc1 : acc0;
        const int col = colBase + wn * 32 + t * 16 + lc;
        const float bv = bias[col];
        #pragma unroll
        for (int j = 0; j < 8; ++j) {
            const int r = rowW + j + 8 * half;
            const float v = acc[j] + bv;
            if (mode == 3) {
                outF[(size_t)r * EE + col] = v;
            } else {
                const int b = r >> 10, n = r & 1023;
                const int h = col >> 6, d = col & 63;
                const size_t idx = (((size_t)b * HH + h) * NN + n) * DD + d;
                outB[idx] = (__bf16)v;
                if (outF) outF[idx] = v;
            }
        }
    }
}

// ---------------------------------------------------------------------------
// TDM: async DMA of one edge tile (d=64 x m=16 x n=16, fp32) global -> LDS.
// D# packed per CDNA5 ISA. pad_enable inserts 4 pad
// dwords after every 64 data dwords -> LDS rows of 68 dwords (bank-friendly).
// This toolchain exposes the 6-arg builtin (extra int32x8 operand, zero-filled).
// ---------------------------------------------------------------------------
#define SE_PAD 68
#define SE_TILE_DW (16 * 16 * SE_PAD)           // 17408 dwords per buffer

__device__ __forceinline__ void tdm_load_edge_tile(const float* gptr, u32 lds_byte_off)
{
    const u64 ga = (u64)(uintptr_t)gptr;
    // group 0: count=1 | lds_addr | global_addr[56:0] | type=2
    v4u g0 = { 1u,
               lds_byte_off,
               (u32)ga,
               ((u32)(ga >> 32) & 0x01FFFFFFu) | (2u << 30) };
    // group 1: data_size=2 (4B), pad_enable=1, pad_interval=5 (64 DW), pad_amount=3 (4 DW)
    //          tensor_dim0=64, tensor_dim1=1024, tile_dim0=64, tile_dim1=16, tile_dim2=16
    //          tensor_dim0_stride=64, tensor_dim1_stride=65536
    v8i g1 = { (int)((2u << 16) | (1u << 20) | (5u << 22) | (3u << 25)),
               (int)(64u << 16),                 // tensor_dim0[15:0] in bits 63:48
               (int)(1024u << 16),               // tensor_dim0 hi=0 | tensor_dim1[15:0]
               (int)(64u << 16),                 // tensor_dim1 hi=0 | tile_dim0=64
               (int)(16u | (16u << 16)),         // tile_dim1=16 | tile_dim2=16
               64,                               // tensor_dim0_stride[31:0]
               0,                                // stride0 hi | tensor_dim1_stride[15:0]=0
               1 };                              // tensor_dim1_stride[47:16] = 1 -> 65536
    // group 2: tensor_dim2=1024, no dim3
    v4i g2 = { 1024, 0, 0, 0 };
    v4i g3 = { 0, 0, 0, 0 };
    v8i g4 = { 0, 0, 0, 0, 0, 0, 0, 0 };        // extra operand of 6-arg form (unused)
    __builtin_amdgcn_tensor_load_to_lds(g0, g1, g2, g3, g4, 0);
}

// ---------------------------------------------------------------------------
// Fused attention: grid (N/16, B), 12 waves = one head each.
// Edge tile streamed by TDM (double-buffered) and shared by all heads.
// Flash-style online softmax; QK^T and PV on bf16 WMMA.
// ---------------------------------------------------------------------------
#define MASKVAL (-3.0e38f)

__global__ void __launch_bounds__(384)
graph_attn_kernel(const __bf16* __restrict__ Qb, const __bf16* __restrict__ Kb,
                  const __bf16* __restrict__ Vb, const float* __restrict__ Qf,
                  const unsigned char* __restrict__ adj,
                  const float* __restrict__ edge, float* __restrict__ attnOut)
{
    extern __shared__ float smem[];
    float*  sE  = smem;                          // 2 x 16*16*68 f32 edge tiles (double buffer)
    float*  sQ  = sE  + 2 * SE_TILE_DW;          // 12*16*64  f32 Q (all heads, this n-block)
    float*  sES = sQ  + HH * 16 * DD;            // 12*16*16  f32 edge scores
    __bf16* sP  = (__bf16*)(sES + HH * 16 * 16); // 12*16*16  bf16 probs staging

    const int tid  = threadIdx.x;
    const int lane = tid & 31;
    const int wave = tid >> 5;                   // == head index h
    const int half = lane >> 4;
    const int lc   = lane & 15;
    const int n0   = blockIdx.x * 16;
    const int b    = blockIdx.y;
    const int h    = wave;

    // cooperatively stage fp32 Q for all heads (edge-dot producer input)
    for (int i = tid; i < HH * 16 * 16; i += 384) {       // float4 granularity
        const int hh  = i >> 8;                            // 256 float4 per head
        const int rem = i & 255;
        const int nn  = rem >> 4;
        const int dq  = (rem & 15) * 4;
        const float4 v = *(const float4*)(Qf + ((((size_t)b * HH + hh) * NN) + n0 + nn) * DD + dq);
        *(float4*)(sQ + (hh * 16 + nn) * DD + dq) = v;
    }

    // per-head Q bf16 A-fragments (d = 0..31 and 32..63)
    v16bf aQ0, aQ1;
    {
        const __bf16* qrow = Qb + (((size_t)b * HH + h) * NN + n0 + lc) * DD;
        #pragma unroll
        for (int j = 0; j < 8; ++j) {
            const int k0 = 16 * half + 2 * j;
            aQ0[2 * j]     = qrow[k0];
            aQ0[2 * j + 1] = qrow[k0 + 1];
            aQ1[2 * j]     = qrow[32 + k0];
            aQ1[2 * j + 1] = qrow[32 + k0 + 1];
        }
    }

    const float scale = 0.03608439182435161f;    // 1/sqrt(E) = 1/sqrt(768)
    float mrow[8], lrow[8];
    #pragma unroll
    for (int j = 0; j < 8; ++j) { mrow[j] = -1.0e30f; lrow[j] = 0.0f; }
    v8f o0 = {}, o1 = {}, o2 = {}, o3 = {};

    const __bf16* Kbh = Kb + ((size_t)b * HH + h) * NN * DD;
    const __bf16* Vbh = Vb + ((size_t)b * HH + h) * NN * DD;

    // kick off first edge tile DMA (one wave issues; EXEC-ignoring TDM op is
    // branched around in other waves via s_cbranch_execz)
    const float* edgeBase = edge + (((size_t)b * NN + n0) * NN) * DD;
    const u32 ldsE0 = (u32)(uintptr_t)(void*)sE;
    if (wave == 0) tdm_load_edge_tile(edgeBase, ldsE0);

    int ib = 0;
    for (int mt = 0; mt < NN; mt += 16) {
        // ---- stream next tile, wait for current (double-buffered TDM) ----
        if (wave == 0) {
            if (mt + 16 < NN) {
                tdm_load_edge_tile(edgeBase + (size_t)(mt + 16) * DD,
                                   ldsE0 + (u32)((ib ^ 1) * SE_TILE_DW * 4));
                __builtin_amdgcn_s_wait_tensorcnt(1);   // current tile landed
            } else {
                __builtin_amdgcn_s_wait_tensorcnt(0);
            }
        }
        __syncthreads();                                 // publish buffer ib
        const float* sEc = sE + ib * SE_TILE_DW;

        // ---- phase 2: edge scores, each edge vector read once for all 12 heads ----
        if (wave < 8) {                                  // 8 waves * 32 lanes = 256 (n,m) pairs
            const int g  = wave * 32 + lane;
            const int nn = g >> 4, mm = g & 15;
            float acc[HH];
            #pragma unroll
            for (int hh = 0; hh < HH; ++hh) acc[hh] = 0.0f;
            const float* ev = sEc + (nn * 16 + mm) * SE_PAD;
            for (int dd = 0; dd < DD; dd += 4) {
                const float4 e4 = *(const float4*)(ev + dd);
                #pragma unroll
                for (int hh = 0; hh < HH; ++hh) {
                    const float4 q4 = *(const float4*)(sQ + (hh * 16 + nn) * DD + dd); // broadcast
                    acc[hh] += e4.x * q4.x + e4.y * q4.y + e4.z * q4.z + e4.w * q4.w;
                }
            }
            #pragma unroll
            for (int hh = 0; hh < HH; ++hh) sES[(hh * 16 + nn) * 16 + mm] = acc[hh];
        }
        __syncthreads();

        // ---- phase 3: per-head QK^T (WMMA) + mask + online softmax + PV (WMMA) ----
        v16bf bK0, bK1;
        {
            const __bf16* krow = Kbh + (size_t)(mt + lc) * DD + 16 * half;
            #pragma unroll
            for (int j = 0; j < 16; ++j) { bK0[j] = krow[j]; bK1[j] = krow[32 + j]; }
        }
        v8f s = {};
        s = __builtin_amdgcn_wmma_f32_16x16x32_bf16(false, aQ0, false, bK0, (short)0, s, false, false);
        s = __builtin_amdgcn_wmma_f32_16x16x32_bf16(false, aQ1, false, bK1, (short)0, s, false, false);

        float p[8];
        #pragma unroll
        for (int j = 0; j < 8; ++j) {
            const int nn = j + 8 * half;
            float sc = (s[j] + sES[(h * 16 + nn) * 16 + lc]) * scale;
            const unsigned char av = adj[((size_t)b * NN + n0 + nn) * NN + mt + lc];
            sc = av ? sc : MASKVAL;
            // row max across the 16 lanes holding this row
            float m = sc;
            m = fmaxf(m, __shfl_xor(m, 1, 32));
            m = fmaxf(m, __shfl_xor(m, 2, 32));
            m = fmaxf(m, __shfl_xor(m, 4, 32));
            m = fmaxf(m, __shfl_xor(m, 8, 32));
            const float mNew = fmaxf(mrow[j], m);
            const float resc = __expf(mrow[j] - mNew);
            const float pv   = __expf(sc - mNew);
            float rs = pv;
            rs += __shfl_xor(rs, 1, 32);
            rs += __shfl_xor(rs, 2, 32);
            rs += __shfl_xor(rs, 4, 32);
            rs += __shfl_xor(rs, 8, 32);
            lrow[j] = lrow[j] * resc + rs;
            mrow[j] = mNew;
            p[j] = pv;
            o0[j] *= resc; o1[j] *= resc; o2[j] *= resc; o3[j] *= resc;
        }

        // probs: C layout -> LDS -> A layout (K padded 16->32 with explicit zeros)
        __bf16* sPh = sP + h * 256;
        #pragma unroll
        for (int j = 0; j < 8; ++j) {
            const int nn = j + 8 * half;
            sPh[nn * 16 + lc] = (__bf16)p[j];
        }
        v16bf aP;
        #pragma unroll
        for (int j = 0; j < 8; ++j) {
            if (j < 4) {
                const int k0 = 2 * j + 8 * half;          // m index 0..15
                aP[2 * j]     = sPh[lc * 16 + k0];
                aP[2 * j + 1] = sPh[lc * 16 + k0 + 1];
            } else {
                aP[2 * j] = (__bf16)0.0f; aP[2 * j + 1] = (__bf16)0.0f;
            }
        }

        #pragma unroll
        for (int t = 0; t < 4; ++t) {                      // V d-chunks of 16
            v16bf bV;
            #pragma unroll
            for (int j = 0; j < 8; ++j) {                   // k = m = 8*half + j real
                bV[j]     = Vbh[(size_t)(mt + 8 * half + j) * DD + t * 16 + lc];
                bV[8 + j] = (__bf16)0.0f;                   // k = 16..31 padded zero
            }
            v8f& o = (t == 0) ? o0 : (t == 1) ? o1 : (t == 2) ? o2 : o3;
            o = __builtin_amdgcn_wmma_f32_16x16x32_bf16(false, aP, false, bV, (short)0, o, false, false);
        }
        __syncthreads();   // close all reads of buffer ib / sES / sP before reuse
        ib ^= 1;
    }

    // ---- epilogue: normalize and write [b][n][h*64+d] fp32 for output GEMM ----
    #pragma unroll
    for (int t = 0; t < 4; ++t) {
        v8f& o = (t == 0) ? o0 : (t == 1) ? o1 : (t == 2) ? o2 : o3;
        #pragma unroll
        for (int j = 0; j < 8; ++j) {
            const int nn = j + 8 * half;
            attnOut[((size_t)b * NN + n0 + nn) * EE + h * DD + t * 16 + lc] = o[j] / lrow[j];
        }
    }
}

// ---------------------------------------------------------------------------
extern "C" void kernel_launch(void* const* d_in, const int* in_sizes, int n_in,
                              void* d_out, int out_size, void* d_ws, size_t ws_size,
                              hipStream_t stream) {
    const float*         X    = (const float*)d_in[0];
    const unsigned char* adj  = (const unsigned char*)d_in[1];  // jax bool = 1 byte
    const float*         edge = (const float*)d_in[2];
    const float* Wq = (const float*)d_in[3];
    const float* bq = (const float*)d_in[4];
    const float* Wk = (const float*)d_in[5];
    const float* bk = (const float*)d_in[6];
    const float* Wv = (const float*)d_in[7];
    const float* bv = (const float*)d_in[8];
    const float* Wo = (const float*)d_in[9];
    const float* bo = (const float*)d_in[10];

    // workspace layout (21 MB total)
    char*   ws = (char*)d_ws;
    __bf16* Qb = (__bf16*)(ws);                       // 3 MB head-major bf16
    __bf16* Kb = (__bf16*)(ws + (size_t)3  * 1024 * 1024);
    __bf16* Vb = (__bf16*)(ws + (size_t)6  * 1024 * 1024);
    float*  Qf = (float*) (ws + (size_t)9  * 1024 * 1024);  // 6 MB head-major fp32
    float*  AO = (float*) (ws + (size_t)15 * 1024 * 1024);  // 6 MB attn out (B,N,E)

    const dim3 gG(32, 12), bG(256);
    wmma_proj_kernel<<<gG, bG, 0, stream>>>(X, Wq, bq, 0, Qb, Qf);
    wmma_proj_kernel<<<gG, bG, 0, stream>>>(X, Wk, bk, 1, Kb, nullptr);
    wmma_proj_kernel<<<gG, bG, 0, stream>>>(X, Wv, bv, 2, Vb, nullptr);

    const size_t shmem = (2 * SE_TILE_DW + HH * 16 * DD + HH * 16 * 16) * sizeof(float)
                       + HH * 16 * 16 * sizeof(__bf16);     // 206,848 B
    graph_attn_kernel<<<dim3(NN / 16, BB), 384, shmem, stream>>>(Qb, Kb, Vb, Qf, adj, edge, AO);

    wmma_proj_kernel<<<gG, bG, 0, stream>>>(AO, Wo, bo, 3, nullptr, (float*)d_out);

    (void)in_sizes; (void)n_in; (void)out_size; (void)ws_size;
}
